// LabelPropagation_11905649344853
// MI455X (gfx1250) — matrix operand order repaired and checked by reading, not compiled, not run on hardware
//
#include <hip/hip_runtime.h>
#include <hip/hip_bf16.h>
#include <math.h>

typedef __attribute__((ext_vector_type(16))) _Float16 v16h;
typedef __attribute__((ext_vector_type(8)))  _Float16 v8h;
typedef __attribute__((ext_vector_type(8)))  float    v8f;

#define F_EPS   2.2204460492503131e-16f
#define F_BNEPS 1e-5f
#define F_ALPHA 0.99f

union U16h { v16h v; _Float16 h[16]; };
union U8h  { v8h  v; _Float16 h[8];  };
union U8f  { v8f  v; float    f[8];  };

// ---------------------------------------------------------------- utilities
__global__ void k_fill_f32(float* p, int n, float val) {
    int i = blockIdx.x * blockDim.x + threadIdx.x;
    if (i < n) p[i] = val;
}

// NCHW f32 (support ++ query) -> halo-padded NHWC f16 ((H+2)x(W+2), interior at +1)
__global__ void k_to_nhwc_pad(const float* __restrict__ sup, const float* __restrict__ qry,
                              _Float16* __restrict__ dst, int Nsup, int C, int H, int W, int Ntot) {
    int idx = blockIdx.x * blockDim.x + threadIdx.x;
    int total = Ntot * H * W * C;
    if (idx >= total) return;
    int ci = idx % C; int t = idx / C;
    int x = t % W; t /= W;
    int y = t % H; t /= H;
    int n = t;
    float v;
    if (n < Nsup) v = sup[(((size_t)n * C + ci) * H + y) * W + x];
    else          v = qry[(((size_t)(n - Nsup) * C + ci) * H + y) * W + x];
    dst[((((size_t)n * (H + 2) + y + 1) * (W + 2)) + (x + 1)) * C + ci] = (_Float16)v;
}

// -------------------------------------------------- weight pre-pack (A frag)
// GEMM K-ordering is kernel-position-major: k' = kpos*CIN + ci  (kpos = 3*(ky+1)+(kx+1))
// A-fragment element e: k' = c*32 + (hi?16:0) + e
template <int CIN>
__global__ void k_pack_w(const float* __restrict__ w, _Float16* __restrict__ apack,
                         int Cout, int nchunks) {
    int tile = blockIdx.x;            // m*nchunks + c
    int m = tile / nchunks, c = tile % nchunks;
    int L = (int)threadIdx.x;
    int hi = L >> 4;
    int row = m * 16 + (L & 15);
    U16h a;
#pragma unroll
    for (int e = 0; e < 16; ++e) {
        int v = e >> 1, pos = e & 1;
        int k = c * 32 + (hi ? 16 : 0) + 2 * v + pos;
        int kpos = k / CIN;
        int ci   = k - kpos * CIN;
        float f = 0.f;
        if (row < Cout && kpos < 9)
            f = w[((row * CIN + ci) * 3 + kpos / 3) * 3 + (kpos % 3)];
        a.h[e] = (_Float16)f;
    }
    *(v16h*)(apack + ((size_t)tile * 32 + L) * 16) = a.v;
}

// per-lane B fragment from halo-padded NHWC activations -- NO bounds checks.
// actn points at image n's padded tensor; (y,x) are logical conv coords; padded
// read row = y + kpos/3, col = x + kpos%3.
template <int CIN>
__device__ inline v16h b_frag_pad(const _Float16* __restrict__ actn, int k0,
                                  int Wi2, int y, int x) {
    U16h b;
    if (CIN == 64) {
        int kpos = k0 >> 6;
        int cib  = k0 & 63;
        int t3  = kpos / 3;
        int kxp = kpos - 3 * t3;
        b.v = *(const v16h*)(actn + (((size_t)(y + t3) * Wi2) + (x + kxp)) * 64 + cib);
    } else {
        int kk = k0;
#pragma unroll
        for (int e = 0; e < 16; ++e) {
            int kpos = kk / CIN;           // constant divisor -> cheap
            int ci   = kk - kpos * CIN;
            float f = 0.f;
            if (kpos < 9)
                f = (float)actn[(((size_t)(y + kpos / 3) * Wi2) + (x + kpos % 3)) * CIN + ci];
            b.h[e] = (_Float16)f;
            ++kk;
        }
    }
    return b.v;
}

// ----------------------------------------- encoder pass A: conv + BN stats
// grid (Nimg, ntiles), block 32 (one wave). 64 output channels = 4 WMMA M-tiles.
template <int CIN>
__global__ void __launch_bounds__(32)
k_conv_stats(const _Float16* __restrict__ act, const _Float16* __restrict__ apack,
             const float* __restrict__ bias, float* __restrict__ sum,
             float* __restrict__ sumsq, int H, int W, int nchunks) {
    int n = blockIdx.x;
    int tile = blockIdx.y;
    int L = (int)threadIdx.x;
    int hi = L >> 4;
    int colL = L & 15;
    int HW = H * W;
    int Wi2 = W + 2;
    int p = tile * 16 + colL;
    bool pvalid = p < HW;
    int y = pvalid ? (p / W) : 0;
    int x = pvalid ? (p - y * W) : 0;
    const _Float16* actn = act + (size_t)n * (H + 2) * Wi2 * CIN;

    v8f zero = {0.f,0.f,0.f,0.f,0.f,0.f,0.f,0.f};
    v8f acc[4];
#pragma unroll
    for (int m = 0; m < 4; ++m) acc[m] = zero;

    int kb = hi ? 16 : 0;
    for (int c = 0; c < nchunks; ++c) {
        v16h b = b_frag_pad<CIN>(actn, c * 32 + kb, Wi2, y, x);
#pragma unroll
        for (int m = 0; m < 4; ++m) {
            v16h a = *(const v16h*)(apack + (((size_t)(m * nchunks + c)) * 32 + L) * 16);
            acc[m] = __builtin_amdgcn_wmma_f32_16x16x32_f16(false, a, false, b,
                                                            (short)0, acc[m], false, false);
        }
    }
#pragma unroll
    for (int m = 0; m < 4; ++m) {
        U8f u; u.v = acc[m];
#pragma unroll
        for (int v = 0; v < 8; ++v) {
            int co = m * 16 + v + hi * 8;
            float val = pvalid ? (u.f[v] + bias[co]) : 0.f;
            float s = val, s2 = val * val;
#pragma unroll
            for (int off = 8; off >= 1; off >>= 1) {
                s  += __shfl_xor(s,  off, 32);
                s2 += __shfl_xor(s2, off, 32);
            }
            if (colL == 0) {
                atomicAdd(&sum[co], s);
                atomicAdd(&sumsq[co], s2);
            }
        }
    }
}

__global__ void k_bn_finalize(const float* __restrict__ sum, const float* __restrict__ sumsq,
                              const float* __restrict__ g, const float* __restrict__ beta,
                              float count, float* __restrict__ scale, float* __restrict__ shift, int C) {
    int c = blockIdx.x * blockDim.x + threadIdx.x;
    if (c >= C) return;
    float m   = sum[c] / count;
    float var = sumsq[c] / count - m * m;
    float sc  = g[c] * rsqrtf(var + F_BNEPS);
    scale[c] = sc;
    shift[c] = beta[c] - m * sc;
}

// fold conv bias into BN shift: shift' = shift + bias*scale
__global__ void k_fold_bias(float* __restrict__ shift, const float* __restrict__ scale,
                            const float* __restrict__ bias, int C) {
    int c = blockIdx.x * blockDim.x + threadIdx.x;
    if (c < C) shift[c] = shift[c] + bias[c] * scale[c];
}

// -------------------- encoder pass B: conv + BN + ReLU + 2x2 maxpool (fused)
// tile = 2 conv rows (2*y0, 2*y0+1) x 8 conv cols (8*strip ..) -> 4 pooled cols
// output NHWC f16, optionally halo-padded ((Hp+2)x(Wp+2), interior at +1)
template <int CIN>
__global__ void __launch_bounds__(32)
k_conv_bnpool(const _Float16* __restrict__ act, const _Float16* __restrict__ apack,
              const float* __restrict__ scale, const float* __restrict__ shift,
              _Float16* __restrict__ out, int H, int W,
              int Hp, int Wp, int nchunks, int nstrips, int halo_out) {
    int n = blockIdx.x;
    int y0 = blockIdx.y / nstrips;
    int strip = blockIdx.y % nstrips;
    int L = (int)threadIdx.x;
    int hi = L >> 4;
    int colL = L & 15;
    int Wi2 = W + 2;
    int y = 2 * y0 + (colL >> 3);       // always < H
    int x = 8 * strip + (colL & 7);     // may exceed W-1 on last strip (garbage, masked)
    const _Float16* actn = act + (size_t)n * (H + 2) * Wi2 * CIN;

    __shared__ float sm[16][16];
    v8f zero = {0.f,0.f,0.f,0.f,0.f,0.f,0.f,0.f};
    v8f acc[4];
#pragma unroll
    for (int m = 0; m < 4; ++m) acc[m] = zero;

    int kb = hi ? 16 : 0;
    for (int c = 0; c < nchunks; ++c) {
        v16h b = b_frag_pad<CIN>(actn, c * 32 + kb, Wi2, y, x);
#pragma unroll
        for (int m = 0; m < 4; ++m) {
            v16h a = *(const v16h*)(apack + (((size_t)(m * nchunks + c)) * 32 + L) * 16);
            acc[m] = __builtin_amdgcn_wmma_f32_16x16x32_f16(false, a, false, b,
                                                            (short)0, acc[m], false, false);
        }
    }
    int Ho = Hp + 2 * halo_out, Wo = Wp + 2 * halo_out;
#pragma unroll
    for (int m = 0; m < 4; ++m) {
        U8f u; u.v = acc[m];
#pragma unroll
        for (int v = 0; v < 8; ++v) {
            int ml = v + hi * 8;
            int co = m * 16 + ml;
            float val = u.f[v] * scale[co] + shift[co];
            sm[ml][colL] = fmaxf(val, 0.f);
        }
        __syncthreads();
#pragma unroll
        for (int t = 0; t < 2; ++t) {
            int o = L + 32 * t;            // 0..63 -> 16 ml x 4 pooled cols
            int ml = o & 15;
            int p  = o >> 4;
            float pv = fmaxf(fmaxf(sm[ml][2 * p], sm[ml][2 * p + 1]),
                             fmaxf(sm[ml][8 + 2 * p], sm[ml][9 + 2 * p]));
            int xp = 4 * strip + p;
            int co = m * 16 + ml;
            if (xp < Wp)
                out[((((size_t)n * Ho + y0 + halo_out) * Wo) + xp + halo_out) * 64 + co]
                    = (_Float16)pv;
        }
        __syncthreads();
    }
}

// ------------------------------------------------ relation head (scalar f32)
// nhwc_in: 1 -> input is [n][H][W][Cin] (the encoder embedding), 0 -> [n][Cin][H][W]
__global__ void k_conv3x3(const _Float16* __restrict__ in, const float* __restrict__ w,
                          const float* __restrict__ bias, float* __restrict__ out,
                          int Nimg, int Cin, int Cout, int H, int W, int nhwc_in) {
    int idx = blockIdx.x * blockDim.x + threadIdx.x;
    int total = Nimg * Cout * H * W;
    if (idx >= total) return;
    int x = idx % W; int t = idx / W;
    int y = t % H;   t /= H;
    int co = t % Cout; int n = t / Cout;
    float acc = bias[co];
    for (int ci = 0; ci < Cin; ++ci)
        for (int ky = -1; ky <= 1; ++ky) {
            int iy = y + ky; if (iy < 0 || iy >= H) continue;
            for (int kx = -1; kx <= 1; ++kx) {
                int ix = x + kx; if (ix < 0 || ix >= W) continue;
                size_t off = nhwc_in
                    ? ((((size_t)n * H + iy) * W + ix) * Cin + ci)
                    : ((((size_t)n * Cin + ci) * H + iy) * W + ix);
                acc += (float)in[off] * w[((co * Cin + ci) * 3 + ky + 1) * 3 + (kx + 1)];
            }
        }
    out[idx] = acc;
}

__global__ void k_chan_stats(const float* __restrict__ x, float* __restrict__ sum,
                             float* __restrict__ sumsq, int Nimg, int C, int HW) {
    int c = blockIdx.x;
    float s = 0.f, s2 = 0.f;
    int tot = Nimg * HW;
    for (int i = threadIdx.x; i < tot; i += blockDim.x) {
        int n = i / HW, p = i - n * HW;
        float v = x[((size_t)n * C + c) * HW + p];
        s += v; s2 += v * v;
    }
    __shared__ float rs[256], rq[256];
    rs[threadIdx.x] = s; rq[threadIdx.x] = s2;
    __syncthreads();
    for (int st = 128; st >= 1; st >>= 1) {
        if ((int)threadIdx.x < st) { rs[threadIdx.x] += rs[threadIdx.x + st];
                                     rq[threadIdx.x] += rq[threadIdx.x + st]; }
        __syncthreads();
    }
    if (threadIdx.x == 0) { sum[c] = rs[0]; sumsq[c] = rq[0]; }
}

// maxpool 2x2 stride 2 pad 1, fused with BN+ReLU (relation head, NCHW)
__global__ void k_bnpool_pad1(const float* __restrict__ x, const float* __restrict__ scale,
                              const float* __restrict__ shift, _Float16* __restrict__ out,
                              int Nimg, int C, int H, int W, int Hp, int Wp) {
    int idx = blockIdx.x * blockDim.x + threadIdx.x;
    int total = Nimg * C * Hp * Wp;
    if (idx >= total) return;
    int px = idx % Wp; int t = idx / Wp;
    int py = t % Hp;   t /= Hp;
    int c = t % C;     int n = t / C;
    float best = 0.f; bool any = false;
    for (int dy = 0; dy < 2; ++dy) {
        int y = 2 * py - 1 + dy; if (y < 0 || y >= H) continue;
        for (int dx = 0; dx < 2; ++dx) {
            int xx = 2 * px - 1 + dx; if (xx < 0 || xx >= W) continue;
            float v = x[(((size_t)n * C + c) * H + y) * W + xx] * scale[c] + shift[c];
            v = fmaxf(v, 0.f);
            best = any ? fmaxf(best, v) : v; any = true;
        }
    }
    out[idx] = (_Float16)best;
}

__global__ void k_fc_sigma(const _Float16* __restrict__ r, const float* __restrict__ w3,
                           const float* __restrict__ b3, const float* __restrict__ w4,
                           const float* __restrict__ b4, float* __restrict__ sigma, int Nimg) {
    int n = blockIdx.x * blockDim.x + threadIdx.x;
    if (n >= Nimg) return;
    float f[4];
    for (int i = 0; i < 4; ++i) f[i] = (float)r[n * 4 + i];
    float sg = b4[0];
    for (int j = 0; j < 8; ++j) {
        float h = b3[j];
        for (int i = 0; i < 4; ++i) h += f[i] * w3[j * 4 + i];
        sg += fmaxf(h, 0.f) * w4[j];
    }
    sigma[n] = sg;
}

// ------------------------------------------------------------ graph section
__global__ void k_make_e(const _Float16* __restrict__ emb, const float* __restrict__ sigma,
                         _Float16* __restrict__ e, int Nimg, int d) {
    int idx = blockIdx.x * blockDim.x + threadIdx.x;
    if (idx >= Nimg * d) return;
    int n = idx / d;
    float v = (float)emb[idx] / (sigma[n] + F_EPS);
    e[idx] = (_Float16)v;
}

__global__ void k_row_sq(const _Float16* __restrict__ e, float* __restrict__ sq, int d) {
    int n = blockIdx.x;
    float s = 0.f;
    for (int j = threadIdx.x; j < d; j += blockDim.x) {
        float v = (float)e[(size_t)n * d + j];
        s += v * v;
    }
    __shared__ float rs[256];
    rs[threadIdx.x] = s; __syncthreads();
    for (int st = 128; st >= 1; st >>= 1) {
        if ((int)threadIdx.x < st) rs[threadIdx.x] += rs[threadIdx.x + st];
        __syncthreads();
    }
    if (threadIdx.x == 0) sq[n] = rs[0];
}

// G = e @ e^T via WMMA, one 16x16 tile per wave.
// e is row-padded to 512 rows (zero-filled) -> fragment loads are guard-free.
__global__ void __launch_bounds__(32)
k_gram_wmma(const _Float16* __restrict__ e, float* __restrict__ G, int N, int d) {
    int it = blockIdx.x, jt = blockIdx.y;
    int L = (int)threadIdx.x;
    int hi = L >> 4, lo = L & 15;
    const _Float16* arow = e + (size_t)(it * 16 + lo) * d + hi * 16;
    const _Float16* bcol = e + (size_t)(jt * 16 + lo) * d + hi * 16;
    int nchunks = d / 32;
    v8f acc = {0.f,0.f,0.f,0.f,0.f,0.f,0.f,0.f};
    for (int c = 0; c < nchunks; ++c) {
        U16h a; U8h a0, a1;
        a0.v = *(const v8h*)(arow + c * 32);
        a1.v = *(const v8h*)(arow + c * 32 + 8);
#pragma unroll
        for (int q = 0; q < 8; ++q) { a.h[q] = a0.h[q]; a.h[8 + q] = a1.h[q]; }
        v16h b = *(const v16h*)(bcol + c * 32);
        __builtin_prefetch(arow + c * 32 + 64, 0, 3);
        acc = __builtin_amdgcn_wmma_f32_16x16x32_f16(false, a.v, false, b,
                                                     (short)0, acc, false, false);
    }
    U8f u; u.v = acc;
#pragma unroll
    for (int v = 0; v < 8; ++v) {
        int i = it * 16 + v + hi * 8;
        int j = jt * 16 + lo;
        if (i < N && j < N) G[(size_t)i * N + j] = u.f[v];
    }
}

__global__ void k_wexp(float* __restrict__ W, const float* __restrict__ sq, int N, float invd) {
    int idx = blockIdx.x * blockDim.x + threadIdx.x;
    if (idx >= N * N) return;
    int i = idx / N, j = idx % N;
    float w = (sq[i] + sq[j] - 2.f * W[idx]) * invd;
    W[idx] = expf(-0.5f * w);
}

__global__ void k_topk_mask(const float* __restrict__ W, float* __restrict__ mask, int N, int K) {
    __shared__ float rowv[512];
    int i = blockIdx.x;
    for (int j = threadIdx.x; j < N; j += 32) rowv[j] = W[(size_t)i * N + j];
    __syncthreads();
    for (int t = 0; t < K; ++t) {
        float best = -1e30f; int bi = 0x7fffffff;
        for (int j = threadIdx.x; j < N; j += 32) {
            float v = rowv[j];
            if (v > best || (v == best && j < bi)) { best = v; bi = j; }
        }
        for (int off = 16; off >= 1; off >>= 1) {
            float ob = __shfl_xor(best, off, 32);
            int   oi = __shfl_xor(bi,   off, 32);
            if (ob > best || (ob == best && oi < bi)) { best = ob; bi = oi; }
        }
        if (threadIdx.x == 0 && bi < N) {
            mask[(size_t)i * N + bi] = 1.f;
            rowv[bi] = -1e30f;
        }
        __syncthreads();
    }
}

__global__ void k_apply_mask(float* __restrict__ W, const float* __restrict__ mask, int N) {
    int idx = blockIdx.x * blockDim.x + threadIdx.x;
    if (idx >= N * N) return;
    int i = idx / N, j = idx % N;
    float ms = mask[(size_t)i * N + j] + mask[(size_t)j * N + i];
    if (!(ms > 0.f)) W[idx] = 0.f;
}

__global__ void k_colsum(const float* __restrict__ W, float* __restrict__ D, int N) {
    int j = blockIdx.x;
    float s = 0.f;
    for (int i = threadIdx.x; i < N; i += blockDim.x) s += W[(size_t)i * N + j];
    __shared__ float rs[256];
    rs[threadIdx.x] = s; __syncthreads();
    for (int st = 128; st >= 1; st >>= 1) {
        if ((int)threadIdx.x < st) rs[threadIdx.x] += rs[threadIdx.x + st];
        __syncthreads();
    }
    if (threadIdx.x == 0) D[j] = rs[0];
}

__global__ void k_dsi(const float* __restrict__ D, float* __restrict__ dsi, int N) {
    int i = blockIdx.x * blockDim.x + threadIdx.x;
    if (i < N) dsi[i] = sqrtf(1.f / (D[i] + F_EPS));
}

// Aug = [ I - alpha*S + eps | y ] padded to 512 columns
__global__ void k_build_aug(const float* __restrict__ W, const float* __restrict__ dsi,
                            const float* __restrict__ slab, float* __restrict__ Aug,
                            int N, int Ns, int C) {
    int idx = blockIdx.x * blockDim.x + threadIdx.x;
    if (idx >= N * 512) return;
    int i = idx / 512, j = idx % 512;
    float v;
    if (j < N) {
        float s = dsi[i] * W[(size_t)i * N + j] * dsi[j];
        v = ((i == j) ? 1.f : 0.f) - F_ALPHA * s + F_EPS;
    } else if (j < N + C) {
        int c = j - N;
        v = (i < Ns) ? slab[i * C + c] : 0.f;
    } else v = 0.f;
    Aug[idx] = v;
}

__global__ void k_elim(float* __restrict__ Aug, int k, int N, int C) {
    int i = blockIdx.x;
    if (i <= k) return;
    __shared__ float fac;
    if (threadIdx.x == 0) fac = Aug[(size_t)i * 512 + k] / Aug[(size_t)k * 512 + k];
    __syncthreads();
    int lim = N + C;
    for (int j = k + threadIdx.x; j < lim; j += blockDim.x)
        Aug[(size_t)i * 512 + j] -= fac * Aug[(size_t)k * 512 + j];
}

__global__ void k_backsub(const float* __restrict__ Aug, float* __restrict__ X, int N, int C) {
    int c = (int)threadIdx.x >> 5;      // blockDim = 32*C
    int lane = (int)threadIdx.x & 31;
    for (int k = N - 1; k >= 0; --k) {
        float s = 0.f;
        for (int j = k + 1 + lane; j < N; j += 32)
            s += Aug[(size_t)k * 512 + j] * X[j * C + c];
        for (int off = 16; off >= 1; off >>= 1) s += __shfl_xor(s, off, 32);
        if (lane == 0)
            X[k * C + c] = (Aug[(size_t)k * 512 + N + c] - s) / Aug[(size_t)k * 512 + k];
        __syncthreads();
    }
}

__global__ void k_finalize(const float* __restrict__ X, const float* __restrict__ slab,
                           const float* __restrict__ qlab, float* __restrict__ out,
                           int N, int Ns, int C) {
    float lsum = 0.f, asum = 0.f;
    for (int r = threadIdx.x; r < N; r += blockDim.x) {
        const float* lab = (r < Ns) ? (slab + r * C) : (qlab + (r - Ns) * C);
        int gt = 0; float bv = lab[0];
        for (int c = 1; c < C; ++c) if (lab[c] > bv) { bv = lab[c]; gt = c; }
        float mx = X[r * C];
        for (int c = 1; c < C; ++c) mx = fmaxf(mx, X[r * C + c]);
        float se = 0.f;
        for (int c = 0; c < C; ++c) se += expf(X[r * C + c] - mx);
        float lse = mx + logf(se);
        lsum += -(X[r * C + gt] - lse);
        if (r >= Ns) {
            int pr = 0; float pv = X[r * C];
            for (int c = 1; c < C; ++c) if (X[r * C + c] > pv) { pv = X[r * C + c]; pr = c; }
            asum += (pr == gt) ? 1.f : 0.f;
        }
    }
    __shared__ float ls[256], as_[256];
    ls[threadIdx.x] = lsum; as_[threadIdx.x] = asum;
    __syncthreads();
    for (int st = 128; st >= 1; st >>= 1) {
        if ((int)threadIdx.x < st) { ls[threadIdx.x] += ls[threadIdx.x + st];
                                     as_[threadIdx.x] += as_[threadIdx.x + st]; }
        __syncthreads();
    }
    if (threadIdx.x == 0) {
        out[0] = ls[0] / (float)N;
        out[1] = as_[0] / (float)(N - Ns);
    }
}

// ------------------------------------------------------------ host plumbing
extern "C" void kernel_launch(void* const* d_in, const int* in_sizes, int n_in,
                              void* d_out, int out_size, void* d_ws, size_t ws_size,
                              hipStream_t stream) {
    const float* support = (const float*)d_in[0];
    const float* slab    = (const float*)d_in[1];
    const float* query   = (const float*)d_in[2];
    const float* qlab    = (const float*)d_in[3];
    const float *ew[4], *eb[4], *eg[4], *ebe[4];
    for (int i = 0; i < 4; ++i) {
        ew[i]  = (const float*)d_in[4 + i * 4 + 0];
        eb[i]  = (const float*)d_in[4 + i * 4 + 1];
        eg[i]  = (const float*)d_in[4 + i * 4 + 2];
        ebe[i] = (const float*)d_in[4 + i * 4 + 3];
    }
    const float* r1w = (const float*)d_in[20]; const float* r1b = (const float*)d_in[21];
    const float* r1g = (const float*)d_in[22]; const float* r1be = (const float*)d_in[23];
    const float* r2w = (const float*)d_in[24]; const float* r2b = (const float*)d_in[25];
    const float* r2g = (const float*)d_in[26]; const float* r2be = (const float*)d_in[27];
    const float* fc3w = (const float*)d_in[28]; const float* fc3b = (const float*)d_in[29];
    const float* fc4w = (const float*)d_in[30]; const float* fc4b = (const float*)d_in[31];

    char* ws = (char*)d_ws;
    size_t off = 0;
    auto alloc = [&](size_t bytes) -> char* {
        char* p = ws + off;
        off = (off + bytes + 255) & ~(size_t)255;
        return p;
    };
    // padded activation sizes (f16 elements):
    // act0p: 500*86*86*3  = 11,094,000   (bufA)
    // act1p: 500*44*44*64 = 61,952,000   (bufB)
    // act2p: 500*23*23*64 = 16,928,000   (bufA)
    // act3p: 500*12*12*64 =  4,608,000   (bufB)
    const size_t szA0 = 11094000, szA2 = 16928000;
    const size_t szB1 = 61952000, szB3 = 4608000;
    _Float16* bufA = (_Float16*)alloc(szA2 * 2 + 65536);   // max(act0p, act2p) + tail pad
    _Float16* bufB = (_Float16*)alloc(szB1 * 2 + 65536);   // act1p + tail pad
    _Float16* emb  = (_Float16*)alloc((size_t)800000 * 2); // 500 x 5 x 5 x 64, compact
    _Float16* ap1  = (_Float16*)alloc((size_t)4 * 1  * 32 * 16 * 2);
    _Float16* ap2  = (_Float16*)alloc((size_t)4 * 18 * 32 * 16 * 2);
    _Float16* ap3  = (_Float16*)alloc((size_t)4 * 18 * 32 * 16 * 2);
    _Float16* ap4  = (_Float16*)alloc((size_t)4 * 18 * 32 * 16 * 2);
    float* sum    = (float*)alloc(64 * 4);
    float* sumsq  = (float*)alloc(64 * 4);
    float* scale  = (float*)alloc(64 * 4);
    float* shift  = (float*)alloc(64 * 4);
    float* rconv1 = (float*)alloc((size_t)800000 * 4);
    _Float16* rpool1 = (_Float16*)alloc((size_t)288000 * 2);
    float* rconv2 = (float*)alloc((size_t)4500 * 4);
    _Float16* rpool2 = (_Float16*)alloc((size_t)2000 * 2);
    float* sigma = (float*)alloc(500 * 4);
    _Float16* ebuf = (_Float16*)alloc((size_t)512 * 1600 * 2); // row-padded to 512
    float* sq   = (float*)alloc(500 * 4);
    float* G    = (float*)alloc((size_t)250000 * 4);
    float* mask = (float*)alloc((size_t)250000 * 4);
    float* Dv   = (float*)alloc(500 * 4);
    float* dsi  = (float*)alloc(500 * 4);
    float* Aug  = (float*)alloc((size_t)500 * 512 * 4);
    float* X    = (float*)alloc((size_t)5000 * 4);

    // zero halo buffers (A for act0p, B for act1p)
    { int n = (int)(szA0 / 2);
      k_fill_f32<<<(n + 255) / 256, 256, 0, stream>>>((float*)bufA, n, 0.f); }
    { int n = (int)(szB1 / 2);
      k_fill_f32<<<(n + 255) / 256, 256, 0, stream>>>((float*)bufB, n, 0.f); }

    // input concat + NCHW f32 -> halo-padded NHWC f16
    { int tot = 500 * 84 * 84 * 3;
      k_to_nhwc_pad<<<(tot + 255) / 256, 256, 0, stream>>>(support, query, bufA,
                                                           50, 3, 84, 84, 500); }

    // pack encoder weights into A-fragment order (kpos-major K)
    k_pack_w<3> <<<4 * 1,  32, 0, stream>>>(ew[0], ap1, 64, 1);
    k_pack_w<64><<<4 * 18, 32, 0, stream>>>(ew[1], ap2, 64, 18);
    k_pack_w<64><<<4 * 18, 32, 0, stream>>>(ew[2], ap3, 64, 18);
    k_pack_w<64><<<4 * 18, 32, 0, stream>>>(ew[3], ap4, 64, 18);

    // encoder blocks
    const _Float16* aps[4] = {ap1, ap2, ap3, ap4};
    const int hs[4]  = {84, 42, 21, 10};
    const int nck[4] = {1, 18, 18, 18};
    _Float16* ins[4]  = {bufA, bufB, bufA, bufB};
    _Float16* outs[4] = {bufB, bufA, bufB, emb};
    const int haloOut[4] = {1, 1, 1, 0};
    const size_t preFill[4] = {0, szA2, szB3, 0};  // zero dest halo buffer before blocks 2,3
    for (int i = 0; i < 4; ++i) {
        int H = hs[i], W = hs[i], HW = H * W;
        int ntiles = (HW + 15) / 16;
        k_fill_f32<<<1, 128, 0, stream>>>(sum, 64, 0.f);
        k_fill_f32<<<1, 128, 0, stream>>>(sumsq, 64, 0.f);
        dim3 g1(500, ntiles);
        if (i == 0)
            k_conv_stats<3> <<<g1, 32, 0, stream>>>(ins[i], aps[i], eb[i], sum, sumsq,
                                                    H, W, nck[i]);
        else
            k_conv_stats<64><<<g1, 32, 0, stream>>>(ins[i], aps[i], eb[i], sum, sumsq,
                                                    H, W, nck[i]);
        k_bn_finalize<<<1, 64, 0, stream>>>(sum, sumsq, eg[i], ebe[i],
                                            500.f * (float)HW, scale, shift, 64);
        k_fold_bias<<<1, 64, 0, stream>>>(shift, scale, eb[i], 64);
        if (preFill[i]) {
            int n = (int)(preFill[i] / 2);
            k_fill_f32<<<(n + 255) / 256, 256, 0, stream>>>((float*)outs[i], n, 0.f);
        }
        int Hp = H / 2, Wp = W / 2, nstrips = (Wp + 3) / 4;
        dim3 g2(500, Hp * nstrips);
        if (i == 0)
            k_conv_bnpool<3> <<<g2, 32, 0, stream>>>(ins[i], aps[i], scale, shift, outs[i],
                                                     H, W, Hp, Wp, nck[i], nstrips, haloOut[i]);
        else
            k_conv_bnpool<64><<<g2, 32, 0, stream>>>(ins[i], aps[i], scale, shift, outs[i],
                                                     H, W, Hp, Wp, nck[i], nstrips, haloOut[i]);
    }
    // emb : 500 x 5 x 5 x 64 (NHWC f16), flat 500x1600

    // relation head (r1 input is NHWC embedding; internals NCHW)
    { int tot = 500 * 64 * 25;
      k_conv3x3<<<(tot + 255) / 256, 256, 0, stream>>>(emb, r1w, r1b, rconv1, 500, 64, 64, 5, 5, 1); }
    k_chan_stats<<<64, 256, 0, stream>>>(rconv1, sum, sumsq, 500, 64, 25);
    k_bn_finalize<<<1, 64, 0, stream>>>(sum, sumsq, r1g, r1be, 500.f * 25.f, scale, shift, 64);
    { int tot = 500 * 64 * 9;
      k_bnpool_pad1<<<(tot + 255) / 256, 256, 0, stream>>>(rconv1, scale, shift, rpool1,
                                                           500, 64, 5, 5, 3, 3); }
    { int tot = 500 * 9;
      k_conv3x3<<<(tot + 255) / 256, 256, 0, stream>>>(rpool1, r2w, r2b, rconv2, 500, 64, 1, 3, 3, 0); }
    k_chan_stats<<<1, 256, 0, stream>>>(rconv2, sum, sumsq, 500, 1, 9);
    k_bn_finalize<<<1, 64, 0, stream>>>(sum, sumsq, r2g, r2be, 500.f * 9.f, scale, shift, 1);
    { int tot = 500 * 4;
      k_bnpool_pad1<<<(tot + 255) / 256, 256, 0, stream>>>(rconv2, scale, shift, rpool2,
                                                           500, 1, 3, 3, 2, 2); }
    k_fc_sigma<<<(500 + 127) / 128, 128, 0, stream>>>(rpool2, fc3w, fc3b, fc4w, fc4b, sigma, 500);

    // graph construction (ebuf zero-filled first -> rows 500..511 stay zero)
    { int n = 512 * 1600 / 2;
      k_fill_f32<<<(n + 255) / 256, 256, 0, stream>>>((float*)ebuf, n, 0.f); }
    { int tot = 800000;
      k_make_e<<<(tot + 255) / 256, 256, 0, stream>>>(emb, sigma, ebuf, 500, 1600); }
    k_row_sq<<<500, 256, 0, stream>>>(ebuf, sq, 1600);
    { dim3 gg(32, 32); k_gram_wmma<<<gg, 32, 0, stream>>>(ebuf, G, 500, 1600); }
    { int tot = 250000;
      k_wexp<<<(tot + 255) / 256, 256, 0, stream>>>(G, sq, 500, 1.f / 1600.f); }
    k_fill_f32<<<(250000 + 255) / 256, 256, 0, stream>>>(mask, 250000, 0.f);
    k_topk_mask<<<500, 32, 0, stream>>>(G, mask, 500, 20);
    { int tot = 250000;
      k_apply_mask<<<(tot + 255) / 256, 256, 0, stream>>>(G, mask, 500); }
    k_colsum<<<500, 256, 0, stream>>>(G, Dv, 500);
    k_dsi<<<(500 + 255) / 256, 256, 0, stream>>>(Dv, dsi, 500);
    { int tot = 500 * 512;
      k_build_aug<<<(tot + 255) / 256, 256, 0, stream>>>(G, dsi, slab, Aug, 500, 50, 10); }

    // Gaussian elimination solve (I - alpha*S + eps) X = y
    for (int k = 0; k < 499; ++k)
        k_elim<<<500, 256, 0, stream>>>(Aug, k, 500, 10);
    k_backsub<<<1, 320, 0, stream>>>(Aug, X, 500, 10);

    k_finalize<<<1, 256, 0, stream>>>(X, slab, qlab, (float*)d_out, 500, 50, 10);
    (void)in_sizes; (void)n_in; (void)out_size; (void)ws_size;
}
